// MultiHeadAttention_33724083208390
// MI455X (gfx1250) — matrix operand order repaired and checked by reading, not compiled, not run on hardware
//
#include <hip/hip_runtime.h>
#include <hip/hip_bf16.h>

// ---------------------------------------------------------------------------
// MHA forward for MI455X (gfx1250), wave32 + WMMA bf16.
// Problem constants from the reference.
// ---------------------------------------------------------------------------
static constexpr int Bb  = 2;
static constexpr int Ss  = 2048;
static constexpr int Dd  = 1024;
static constexpr int Hh  = 16;
static constexpr int DKk = 64;
static constexpr float LN_EPS = 1e-5f;

typedef float  v8f   __attribute__((ext_vector_type(8)));
typedef __bf16 v16bf __attribute__((ext_vector_type(16)));
typedef __bf16 v2bf  __attribute__((ext_vector_type(2)));
typedef unsigned short ushort_t;

// fp32 -> bf16 via HW conversion (v_cvt_pk_bf16_f32 class), not manual bit ops.
__device__ __forceinline__ unsigned pk2(float a, float b) {
#if __has_builtin(__builtin_amdgcn_cvt_pk_bf16_f32)
  union { v2bf v; unsigned u; } c;
  c.v = __builtin_amdgcn_cvt_pk_bf16_f32(a, b);
  return c.u;
#else
  union { v2bf v; unsigned u; } c;
  c.v[0] = (__bf16)a; c.v[1] = (__bf16)b;
  return c.u;
#endif
}
__device__ __forceinline__ ushort_t f2bf_h(float f) {
  union { __bf16 h; ushort_t u; } c;
  c.h = (__bf16)f;
  return c.u;
}

// ---------------------------------------------------------------------------
// Unified WMMA GEMM:  C[m,n] = alpha * sum_k A[m,k] * B[n,k]  (+ resid)
//   - A is MxK row-major (fp32 or bf16), B is NxK row-major ("W[out,in]").
//   - Both are converted to bf16 while staging through LDS.
//   - Batched over gridDim.z = b*batchH + h with independent b/h strides.
//   - EPI: 0 = fp32 store (+optional residual add), 1 = bf16 store.
// Block: 256 threads (8 wave32), 128x64 output tile, BK = 32 (= WMMA K).
// Each wave: 32x32 tile = 2x2 v_wmma_f32_16x16x32_bf16 accumulators.
// ---------------------------------------------------------------------------
template<typename TA, typename TB, int EPI>
__global__ __launch_bounds__(256) void gemm_wmma_kernel(
    const void* __restrict__ Av, long lda, long sAb, long sAh,
    const void* __restrict__ Bv, long ldb, long sBb, long sBh,
    void*       __restrict__ Cv, long ldc, long sCb, long sCh,
    int M, int N, int K, int batchH, float alpha,
    const float* __restrict__ resid)
{
  constexpr int BM = 128, BN = 64, BK = 32;
  constexpr int RS = 40;  // LDS row stride (bf16 elems): 80B = 16B-aligned, staggers banks
  __shared__ ushort_t As[BM * RS];   // 10 KB
  __shared__ ushort_t Bs[BN * RS];   //  5 KB

  const int tid  = threadIdx.x;
  const int lane = tid & 31;
  const int w    = tid >> 5;
  const int wm   = w & 3;        // 4 waves along M (4*32 = 128)
  const int wn   = w >> 2;       // 2 waves along N (2*32 = 64)

  const int zb = blockIdx.z / batchH;
  const int zh = blockIdx.z % batchH;
  const TA* A  = (const TA*)Av + (long)zb * sAb + (long)zh * sAh;
  const TB* Bm = (const TB*)Bv + (long)zb * sBb + (long)zh * sBh;
  const long cOff = (long)zb * sCb + (long)zh * sCh;

  const long rowBase = (long)blockIdx.y * BM;
  const long colBase = (long)blockIdx.x * BN;

  // Staging coords: A tile 128x32 -> 16 elems/thread; B tile 64x32 -> 8 elems/thread
  const int arow = tid >> 1, acol = (tid & 1) * 16;
  const int brow = tid >> 2, bcol = (tid & 3) * 8;

  v8f acc[2][2];
  #pragma unroll
  for (int i = 0; i < 2; i++)
    #pragma unroll
    for (int j = 0; j < 2; j++)
      #pragma unroll
      for (int e = 0; e < 8; e++) acc[i][j][e] = 0.0f;

  for (int k0 = 0; k0 < K; k0 += BK) {
    // ---- stage A tile (HW-convert to bf16 if fp32) ----
    {
      const long off = (rowBase + arow) * lda + k0 + acol;
      ushort_t* dst = &As[arow * RS + acol];
      if constexpr (sizeof(TA) == 4) {
        const float4* s = (const float4*)((const float*)A + off);
        float4 f0 = s[0], f1 = s[1], f2 = s[2], f3 = s[3];
        uint4 u0 = { pk2(f0.x,f0.y), pk2(f0.z,f0.w), pk2(f1.x,f1.y), pk2(f1.z,f1.w) };
        uint4 u1 = { pk2(f2.x,f2.y), pk2(f2.z,f2.w), pk2(f3.x,f3.y), pk2(f3.z,f3.w) };
        *(uint4*)dst = u0;  *(uint4*)(dst + 8) = u1;
      } else {
        const uint4* s = (const uint4*)((const ushort_t*)A + off);
        *(uint4*)dst = s[0];  *(uint4*)(dst + 8) = s[1];
      }
    }
    // ---- stage B tile ----
    {
      const long off = (colBase + brow) * ldb + k0 + bcol;
      ushort_t* dst = &Bs[brow * RS + bcol];
      if constexpr (sizeof(TB) == 4) {
        const float4* s = (const float4*)((const float*)Bm + off);
        float4 f0 = s[0], f1 = s[1];
        uint4 u0 = { pk2(f0.x,f0.y), pk2(f0.z,f0.w), pk2(f1.x,f1.y), pk2(f1.z,f1.w) };
        *(uint4*)dst = u0;
      } else {
        *(uint4*)dst = *(const uint4*)((const ushort_t*)Bm + off);
      }
    }
    __syncthreads();

    // prefetch next K tile into caches (global_prefetch_b8) while we compute
    if (k0 + BK < K) {
      __builtin_prefetch((const void*)(A  + (rowBase + arow) * lda + k0 + BK + acol), 0, 1);
      __builtin_prefetch((const void*)(Bm + (colBase + brow) * ldb + k0 + BK + bcol), 0, 1);
    }

    // ---- load fragments per ISA 7.12.2 VGPR layouts ----
    const int half = lane >> 4;
    const int l15  = lane & 15;
    union Frag { uint4 u[2]; v16bf v; };
    Frag a[2], b[2];
    #pragma unroll
    for (int mt = 0; mt < 2; mt++) {
      // A 16x32: lanes 0-15 hold K {0..7,16..23}; lanes 16-31 K {8..15,24..31}
      const ushort_t* p = &As[(wm * 32 + mt * 16 + l15) * RS + half * 8];
      a[mt].u[0] = *(const uint4*)p;
      a[mt].u[1] = *(const uint4*)(p + 16);
    }
    #pragma unroll
    for (int nt = 0; nt < 2; nt++) {
      // B 32x16: lanes 0-15 hold K {0..7,16..23}; lanes 16-31 K {8..15,24..31}
      const ushort_t* p = &Bs[(wn * 32 + nt * 16 + l15) * RS + half * 8];
      b[nt].u[0] = *(const uint4*)p;
      b[nt].u[1] = *(const uint4*)(p + 16);
    }

    #pragma unroll
    for (int mt = 0; mt < 2; mt++)
      #pragma unroll
      for (int nt = 0; nt < 2; nt++)
        acc[mt][nt] = __builtin_amdgcn_wmma_f32_16x16x32_bf16(
            false, a[mt].v, false, b[nt].v, (short)0, acc[mt][nt], false, false);

    __syncthreads();
  }

  // ---- epilogue: C/D layout -> VGPR r holds (M = r + 8*half, N = lane&15) ----
  #pragma unroll
  for (int mt = 0; mt < 2; mt++) {
    #pragma unroll
    for (int nt = 0; nt < 2; nt++) {
      const long gr0 = rowBase + wm * 32 + mt * 16 + 8 * (lane >> 4);
      const long gc  = colBase + wn * 32 + nt * 16 + (lane & 15);
      #pragma unroll
      for (int r = 0; r < 8; r++) {
        const long idx = (gr0 + r) * ldc + gc;
        float v = alpha * acc[mt][nt][r];
        if constexpr (EPI == 0) {
          if (resid) v += resid[idx];
          ((float*)Cv)[cOff + idx] = v;
        } else {
          ((ushort_t*)Cv)[cOff + idx] = f2bf_h(v);
        }
      }
    }
  }
}

// ---------------------------------------------------------------------------
// Row softmax with mask, in place on the fp32 attn region of d_out.
// One 256-thread block per row (B*H*S rows, S=2048 -> 8 elems/thread).
// ---------------------------------------------------------------------------
__global__ __launch_bounds__(256) void softmax_mask_kernel(
    float* __restrict__ attn, const unsigned char* __restrict__ mask)
{
  __shared__ float red[256];
  const int tid  = threadIdx.x;
  const long row = blockIdx.x;                 // b*H*S + h*S + q
  const long base = row * (long)Ss;
  const int bi = (int)(row / ((long)Hh * Ss));
  const int qi = (int)(row % Ss);
  const unsigned char* mrow = mask + ((long)bi * Ss + qi) * (long)Ss;

  float v[8];
  float mx = -3.0e38f;
  #pragma unroll
  for (int i = 0; i < 8; i++) {
    const int c = tid + i * 256;
    float s = attn[base + c];
    if (mrow[c]) s = -1e9f;
    v[i] = s; mx = fmaxf(mx, s);
  }
  red[tid] = mx; __syncthreads();
  for (int s = 128; s > 0; s >>= 1) { if (tid < s) red[tid] = fmaxf(red[tid], red[tid + s]); __syncthreads(); }
  mx = red[0]; __syncthreads();

  float sum = 0.0f;
  #pragma unroll
  for (int i = 0; i < 8; i++) { v[i] = __expf(v[i] - mx); sum += v[i]; }
  red[tid] = sum; __syncthreads();
  for (int s = 128; s > 0; s >>= 1) { if (tid < s) red[tid] += red[tid + s]; __syncthreads(); }
  const float inv = 1.0f / red[0];

  #pragma unroll
  for (int i = 0; i < 8; i++) attn[base + tid + i * 256] = v[i] * inv;
}

// ---------------------------------------------------------------------------
// bf16 transpose: V [B, S, D] -> Vt [B, D, S] (so the context GEMM's B-loader
// reads contiguous K like every other GEMM). 32x32 LDS tiles, blockDim (32,8).
// ---------------------------------------------------------------------------
__global__ __launch_bounds__(256) void transpose_bf16_kernel(
    const ushort_t* __restrict__ in, ushort_t* __restrict__ outp)
{
  __shared__ ushort_t tile[32][33];
  const int b  = blockIdx.z;
  const int c0 = blockIdx.x * 32;   // feature coord
  const int s0 = blockIdx.y * 32;   // sequence coord
  #pragma unroll
  for (int i = 0; i < 4; i++) {
    const int lr = threadIdx.y + i * 8;
    tile[lr][threadIdx.x] = in[((long)b * Ss + s0 + lr) * Dd + c0 + threadIdx.x];
  }
  __syncthreads();
  #pragma unroll
  for (int i = 0; i < 4; i++) {
    const int lr = threadIdx.y + i * 8;
    outp[((long)b * Dd + c0 + lr) * Ss + s0 + threadIdx.x] = tile[threadIdx.x][lr];
  }
}

// ---------------------------------------------------------------------------
// LayerNorm over last dim (D=1024), one 256-thread block per row.
// ---------------------------------------------------------------------------
__global__ __launch_bounds__(256) void layernorm_kernel(
    const float* __restrict__ x, const float* __restrict__ g,
    const float* __restrict__ be, float* __restrict__ outp)
{
  __shared__ float r1[256], r2[256];
  const int tid = threadIdx.x;
  const long base = (long)blockIdx.x * Dd;
  float v[4]; float s = 0.0f, ss = 0.0f;
  #pragma unroll
  for (int i = 0; i < 4; i++) {
    const float t = x[base + tid + i * 256];
    v[i] = t; s += t; ss += t * t;
  }
  r1[tid] = s; r2[tid] = ss; __syncthreads();
  for (int k = 128; k > 0; k >>= 1) {
    if (tid < k) { r1[tid] += r1[tid + k]; r2[tid] += r2[tid + k]; }
    __syncthreads();
  }
  const float mu   = r1[0] * (1.0f / Dd);
  const float var  = r2[0] * (1.0f / Dd) - mu * mu;
  const float rstd = rsqrtf(var + LN_EPS);
  #pragma unroll
  for (int i = 0; i < 4; i++) {
    const int c = tid + i * 256;
    outp[base + c] = (v[i] - mu) * rstd * g[c] + be[c];
  }
}

// ---------------------------------------------------------------------------
// Host orchestration (graph-capture safe: only kernel launches on `stream`).
// ---------------------------------------------------------------------------
extern "C" void kernel_launch(void* const* d_in, const int* in_sizes, int n_in,
                              void* d_out, int out_size, void* d_ws, size_t ws_size,
                              hipStream_t stream)
{
  (void)in_sizes; (void)n_in; (void)out_size; (void)ws_size;

  const float* input_Q = (const float*)d_in[0];
  const float* input_K = (const float*)d_in[1];
  const float* input_V = (const float*)d_in[2];
  const unsigned char* attn_mask = (const unsigned char*)d_in[3];
  const float* fc0_w = (const float*)d_in[4];
  const float* wq_w  = (const float*)d_in[5];
  const float* wk_w  = (const float*)d_in[6];
  const float* wv_w  = (const float*)d_in[7];
  const float* fc_w  = (const float*)d_in[8];
  const float* ln_g  = (const float*)d_in[9];
  const float* ln_b  = (const float*)d_in[10];

  // workspace layout (~76 MB)
  char* ws = (char*)d_ws;
  size_t off = 0;
  auto alloc = [&](size_t bytes) -> void* {
    void* p = ws + off; off += (bytes + 255) & ~(size_t)255; return p;
  };
  float*    residual = (float*)alloc((size_t)Bb * Ss * Dd * 4);
  float*    xbuf     = (float*)alloc((size_t)Bb * Ss * Dd * 4);
  ushort_t* Qbf      = (ushort_t*)alloc((size_t)Bb * Ss * Dd * 2);
  ushort_t* Kbf      = (ushort_t*)alloc((size_t)Bb * Ss * Dd * 2);
  ushort_t* Vbf      = (ushort_t*)alloc((size_t)Bb * Ss * Dd * 2);
  ushort_t* Vt       = (ushort_t*)alloc((size_t)Bb * Ss * Dd * 2);
  ushort_t* ctx      = (ushort_t*)alloc((size_t)Bb * Ss * Dd * 2);

  float* outp = (float*)d_out;
  float* attn = outp + (size_t)Bb * Ss * Dd;      // tuple order: (out, attn)

  const long SD  = (long)Ss * Dd;                 // per-batch token-feature stride
  const long SS2 = (long)Ss * Ss;                 // per-head attn stride

  const dim3 thr(256);
  const dim3 gProj(Dd / 64, (Bb * Ss) / 128, 1);

  // 1) residual = input_Q @ fc0_w^T  (fp32 out)
  gemm_wmma_kernel<float, float, 0><<<gProj, thr, 0, stream>>>(
      input_Q, Dd, 0, 0, fc0_w, Dd, 0, 0, residual, Dd, 0, 0,
      Bb * Ss, Dd, Dd, 1, 1.0f, nullptr);

  // 2) Q/K/V projections -> bf16 [B,S,H*64]
  gemm_wmma_kernel<float, float, 1><<<gProj, thr, 0, stream>>>(
      input_Q, Dd, 0, 0, wq_w, Dd, 0, 0, Qbf, Dd, 0, 0,
      Bb * Ss, Dd, Dd, 1, 1.0f, nullptr);
  gemm_wmma_kernel<float, float, 1><<<gProj, thr, 0, stream>>>(
      input_K, Dd, 0, 0, wk_w, Dd, 0, 0, Kbf, Dd, 0, 0,
      Bb * Ss, Dd, Dd, 1, 1.0f, nullptr);
  gemm_wmma_kernel<float, float, 1><<<gProj, thr, 0, stream>>>(
      input_V, Dd, 0, 0, wv_w, Dd, 0, 0, Vbf, Dd, 0, 0,
      Bb * Ss, Dd, Dd, 1, 1.0f, nullptr);

  // 3) Vt[b, d, s] = V[b, s, d]
  transpose_bf16_kernel<<<dim3(Dd / 32, Ss / 32, Bb), dim3(32, 8), 0, stream>>>(Vbf, Vt);

  // 4) scores = (Q @ K^T) / sqrt(64), per (b,h), fp32 straight into d_out attn region
  gemm_wmma_kernel<ushort_t, ushort_t, 0><<<dim3(Ss / 64, Ss / 128, Bb * Hh), thr, 0, stream>>>(
      Qbf, Dd, SD, DKk, Kbf, Dd, SD, DKk, attn, Ss, (long)Hh * SS2, SS2,
      Ss, Ss, DKk, Hh, 0.125f, nullptr);

  // 5) masked softmax in place (also produces the final attn output)
  softmax_mask_kernel<<<dim3(Bb * Hh * Ss), thr, 0, stream>>>(attn, attn_mask);

  // 6) context = attn @ V  (A = fp32 attn converted in-register, B = Vt bf16)
  gemm_wmma_kernel<float, ushort_t, 1><<<dim3(1, Ss / 128, Bb * Hh), thr, 0, stream>>>(
      attn, Ss, (long)Hh * SS2, SS2, Vt, Ss, (long)Dd * Ss, (long)DKk * Ss,
      ctx, Dd, SD, DKk, Ss, DKk, Ss, Hh, 1.0f, nullptr);

  // 7) x = context @ fc_w^T + residual  (fp32)
  gemm_wmma_kernel<ushort_t, float, 0><<<gProj, thr, 0, stream>>>(
      ctx, Dd, 0, 0, fc_w, Dd, 0, 0, xbuf, Dd, 0, 0,
      Bb * Ss, Dd, Dd, 1, 1.0f, residual);

  // 8) LayerNorm -> d_out[0 : B*S*D]
  layernorm_kernel<<<dim3(Bb * Ss), thr, 0, stream>>>(xbuf, ln_g, ln_b, outp);
}
